// PAF_59416577573149
// MI455X (gfx1250) — matrix operand order repaired and checked by reading, not compiled, not run on hardware
//
#include <hip/hip_runtime.h>

// ---------------------------------------------------------------------------
// SAGAN self-attention, fused flash-attention formulation for gfx1250 (MI455X)
//   B=4, N=4096, C=256.  bf16 WMMA (v_wmma_f32_16x16x32_bf16), f32 accumulate.
//   K/V tiles staged block-wide into LDS with global_load_async_to_lds_b128
//   (ASYNCcnt pipeline), 4x reduction in L2 read traffic vs per-wave loads.
// ---------------------------------------------------------------------------

#define B_SZ   4
#define N_SEQ  4096
#define C_DIM  256

typedef __bf16 bf16_t;
typedef __attribute__((ext_vector_type(16))) __bf16 v16bf;
typedef __attribute__((ext_vector_type(8)))  __bf16 v8bf;
typedef __attribute__((ext_vector_type(8)))  float  v8f;

// ---- fragment loaders (layouts per CDNA5 ISA 7.12.2) ----------------------
// A-matrix 16x32 bf16: lane l holds row m=l&15; k0=(l>>4)*8; two contiguous
// 8-element (16B) chunks at k0 and k0+16.  Source must be row-major [m][k].
__device__ __forceinline__ v16bf load_fragA(const bf16_t* base, int ld, int lane) {
  const int m  = lane & 15;
  const int k0 = (lane >> 4) * 8;
  const bf16_t* p = base + (long)m * ld + k0;
  v8bf lo = *(const v8bf*)(p);
  v8bf hi = *(const v8bf*)(p + 16);
  v16bf r;
#pragma unroll
  for (int i = 0; i < 8; ++i) { r[i] = lo[i]; r[i + 8] = hi[i]; }
  return r;
}

// B-matrix 32x16 bf16 (computing X * Yt with Y stored row-major [n][k]):
// lane l holds column n=l&15; kb=(l>>4)*8; two contiguous 8-element chunks at kb and kb+16.
__device__ __forceinline__ v16bf load_fragB(const bf16_t* base, int ld, int lane) {
  const int n  = lane & 15;
  const int kb = (lane >> 4) * 8;
  const bf16_t* p = base + (long)n * ld + kb;
  v8bf lo = *(const v8bf*)(p);
  v8bf hi = *(const v8bf*)(p + 16);
  v16bf r;
#pragma unroll
  for (int i = 0; i < 8; ++i) { r[i] = lo[i]; r[i + 8] = hi[i]; }
  return r;
}

// ---- CDNA5 async DMA: global -> LDS, tracked by ASYNCcnt ------------------
__device__ __forceinline__ void async_load_b128(unsigned lds_addr, const void* gaddr) {
  asm volatile("global_load_async_to_lds_b128 %0, %1, off"
               :: "v"(lds_addr), "v"(gaddr) : "memory");
}

// ---- kernel 0: f32 -> bf16 cast -------------------------------------------
__global__ void PAF_cast_kernel(const float* __restrict__ src,
                                bf16_t* __restrict__ dst, int n) {
  int i = blockIdx.x * blockDim.x + threadIdx.x;
  int stride = gridDim.x * blockDim.x;
  for (; i < n; i += stride) dst[i] = (bf16_t)src[i];
}

// ---- kernel 1: QKV projection (WMMA GEMM) ---------------------------------
// q[b][n][c], k[b][n][c] row-major bf16; v stored transposed vT[b][c][n].
__global__ __launch_bounds__(256)
void PAF_proj_kernel(const bf16_t* __restrict__ xbf,
                     const bf16_t* __restrict__ wq,
                     const bf16_t* __restrict__ wk,
                     const bf16_t* __restrict__ wv,
                     const float* __restrict__ bq,
                     const float* __restrict__ bk,
                     const float* __restrict__ bv,
                     bf16_t* __restrict__ qbf,
                     bf16_t* __restrict__ kbf,
                     bf16_t* __restrict__ vTbf) {
  const int lane = threadIdx.x & 31;
  const int wave = threadIdx.x >> 5;
  const int tile = blockIdx.x * 8 + wave;          // 49152 tiles total

  const int TILES_PER_MAT = B_SZ * (N_SEQ / 16) * (C_DIM / 16); // 16384
  const int mat  = tile / TILES_PER_MAT;           // 0=q 1=k 2=v
  int rem        = tile % TILES_PER_MAT;
  const int bidx = rem / ((N_SEQ / 16) * (C_DIM / 16));
  rem            = rem % ((N_SEQ / 16) * (C_DIM / 16));
  const int qt   = rem / (C_DIM / 16);
  const int ct   = rem % (C_DIM / 16);

  const bf16_t* W    = (mat == 0) ? wq : (mat == 1) ? wk : wv;
  const float*  bias = (mat == 0) ? bq : (mat == 1) ? bk : bv;

  const bf16_t* A  = xbf + ((long)bidx * N_SEQ + qt * 16) * C_DIM;
  const bf16_t* Bm = W + (long)ct * 16 * C_DIM;    // W rows = output channels

  v8f acc = {0.f, 0.f, 0.f, 0.f, 0.f, 0.f, 0.f, 0.f};
#pragma unroll
  for (int kc = 0; kc < C_DIM; kc += 32) {
    v16bf a = load_fragA(A + kc, C_DIM, lane);
    v16bf b = load_fragB(Bm + kc, C_DIM, lane);
    acc = __builtin_amdgcn_wmma_f32_16x16x32_bf16(false, a, false, b,
                                                  (short)0, acc, false, false);
  }

  const int ncol  = lane & 15;
  const int mrow0 = (lane >> 4) * 8;
  const float bval = bias[ct * 16 + ncol];

  if (mat == 2) {  // vT[b][c][n]: per lane 8 consecutive n positions
    bf16_t* o = vTbf + ((long)bidx * C_DIM + ct * 16 + ncol) * N_SEQ
                     + qt * 16 + mrow0;
#pragma unroll
    for (int r = 0; r < 8; ++r) o[r] = (bf16_t)(acc[r] + bval);
  } else {         // q/k[b][n][c]
    bf16_t* o = ((mat == 0) ? qbf : kbf)
              + ((long)bidx * N_SEQ + qt * 16 + mrow0) * C_DIM + ct * 16 + ncol;
#pragma unroll
    for (int r = 0; r < 8; ++r) o[(long)r * C_DIM] = (bf16_t)(acc[r] + bval);
  }
}

// ---- kernel 2: fused flash attention with async K/V staging ---------------
// One wave = one 16-query tile; 4 waves/block share one batch's K/V stream.
#define ATT_WAVES 4
#define KTILE     32                 // keys per iteration
#define ITERS     (N_SEQ / KTILE)    // 128

__global__ __launch_bounds__(ATT_WAVES * 32)
void PAF_attn_kernel(const bf16_t* __restrict__ qbf,
                     const bf16_t* __restrict__ kbf,
                     const bf16_t* __restrict__ vTbf,
                     const float* __restrict__ x,
                     const float* __restrict__ gamma_p,
                     float* __restrict__ out) {
  // K: [key][c] rows (32 x 256 bf16 = 16KB), double buffered.
  // V: [c][key] rows (256 x 32 bf16 = 16KB), single buffered (issued at the
  //    end of the previous iteration -> waited on before next S-GEMM).
  __shared__ __align__(16) bf16_t shK[2][KTILE * C_DIM];   // 32 KB
  __shared__ __align__(16) bf16_t shV[C_DIM * KTILE];      // 16 KB
  __shared__ __align__(16) bf16_t shP[ATT_WAVES][16 * 32]; //  4 KB

  const int tid  = threadIdx.x;       // 0..127
  const int lane = tid & 31;
  const int wave = tid >> 5;
  const int tile = blockIdx.x * ATT_WAVES + wave;  // 1024 tiles
  const int b  = tile >> 8;           // uniform per block (64 blocks / batch)
  const int qt = tile & 255;
  const float g = gamma_p[0];

  const bf16_t* Kg = kbf  + (long)b * N_SEQ * C_DIM;
  const bf16_t* Vt = vTbf + (long)b * C_DIM * N_SEQ;

  const unsigned kbase = (unsigned)(uintptr_t)&shK[0][0];
  const unsigned vbase = (unsigned)(uintptr_t)&shV[0];
  const unsigned KBYTES = KTILE * C_DIM * 2;       // 16384

  // Each thread DMAs 8 K-chunks (B128) per stage: K tile is contiguous 16KB.
  auto stageK = [&](int buf, int jb) {
    const char* src = (const char*)(Kg + (long)jb * C_DIM);
    unsigned dst = kbase + (unsigned)buf * KBYTES;
#pragma unroll
    for (int i = 0; i < 8; ++i) {
      int chunk = tid + i * 128;                   // 1024 chunks of 16B
      async_load_b128(dst + chunk * 16, src + chunk * 16);
    }
  };
  // Each thread DMAs 8 V-chunks: row c of vT -> 64B (4 chunks per row).
  auto stageV = [&](int jb) {
#pragma unroll
    for (int i = 0; i < 8; ++i) {
      int chunk = tid + i * 128;
      int c = chunk >> 2, q = chunk & 3;
      async_load_b128(vbase + chunk * 16,
                      (const char*)(Vt + (long)c * N_SEQ + jb) + q * 16);
    }
  };

  // ---- Q tile lives in registers: 8 A-fragments (16 x 256 bf16) ----
  const bf16_t* Qg = qbf + ((long)b * N_SEQ + qt * 16) * C_DIM;
  v16bf qfrag[8];
#pragma unroll
  for (int kc = 0; kc < 8; ++kc)
    qfrag[kc] = load_fragA(Qg + kc * 32, C_DIM, lane);

  // online-softmax state: lane half owns rows m = (lane>>4)*8 + r
  float m_i[8], l_i[8];
#pragma unroll
  for (int r = 0; r < 8; ++r) { m_i[r] = -3.0e38f; l_i[r] = 0.f; }
  v8f o_acc[16];
#pragma unroll
  for (int t = 0; t < 16; ++t)
#pragma unroll
    for (int r = 0; r < 8; ++r) o_acc[t][r] = 0.f;

  const int ncol  = lane & 15;
  const int mrow0 = (lane >> 4) * 8;

  stageK(0, 0);
  stageV(0);

  for (int it = 0; it < ITERS; ++it) {
    const int cur = it & 1;
    const int jb  = it * KTILE;

    if (it + 1 < ITERS) {
      stageK(1 - cur, jb + KTILE);   // overlaps this iteration's compute
      asm volatile("s_wait_asynccnt 0x8" ::: "memory"); // only K(it+1) pending
    } else {
      asm volatile("s_wait_asynccnt 0x0" ::: "memory");
    }
    __syncthreads();                 // all waves' K[cur] + V(it) DMAs visible

    // ---- S = Q * K^T : two 16x16 f32 tiles from LDS-staged K ----
    v8f s0 = {0.f, 0.f, 0.f, 0.f, 0.f, 0.f, 0.f, 0.f};
    v8f s1 = {0.f, 0.f, 0.f, 0.f, 0.f, 0.f, 0.f, 0.f};
    const bf16_t* Kc = &shK[cur][0];
#pragma unroll
    for (int kc = 0; kc < 8; ++kc) {
      v16bf b0 = load_fragB(Kc + kc * 32, C_DIM, lane);            // keys 0-15
      s0 = __builtin_amdgcn_wmma_f32_16x16x32_bf16(false, qfrag[kc], false, b0,
                                                   (short)0, s0, false, false);
      v16bf b1 = load_fragB(Kc + 16 * C_DIM + kc * 32, C_DIM, lane); // 16-31
      s1 = __builtin_amdgcn_wmma_f32_16x16x32_bf16(false, qfrag[kc], false, b1,
                                                   (short)0, s1, false, false);
    }

    // ---- online softmax (row stats live in one 16-lane half) ----
    float mnew[8], alpha[8], p0[8], p1[8];
#pragma unroll
    for (int r = 0; r < 8; ++r) {
      float t = fmaxf(s0[r], s1[r]);
      t = fmaxf(t, __shfl_xor(t, 1));
      t = fmaxf(t, __shfl_xor(t, 2));
      t = fmaxf(t, __shfl_xor(t, 4));
      t = fmaxf(t, __shfl_xor(t, 8));
      mnew[r]  = fmaxf(m_i[r], t);
      alpha[r] = __expf(m_i[r] - mnew[r]);
      m_i[r]   = mnew[r];
    }
#pragma unroll
    for (int r = 0; r < 8; ++r) {
      p0[r] = __expf(s0[r] - mnew[r]);
      p1[r] = __expf(s1[r] - mnew[r]);
      float u = p0[r] + p1[r];
      u += __shfl_xor(u, 1);
      u += __shfl_xor(u, 2);
      u += __shfl_xor(u, 4);
      u += __shfl_xor(u, 8);
      l_i[r] = l_i[r] * alpha[r] + u;
    }
#pragma unroll
    for (int t = 0; t < 16; ++t)
#pragma unroll
      for (int r = 0; r < 8; ++r) o_acc[t][r] *= alpha[r];

    // ---- reshape P (C/D layout) -> A layout through LDS ----
    __syncthreads();                 // WAR vs previous iteration's frag reads
#pragma unroll
    for (int r = 0; r < 8; ++r) {
      shP[wave][(mrow0 + r) * 32 + ncol]      = (bf16_t)p0[r];
      shP[wave][(mrow0 + r) * 32 + 16 + ncol] = (bf16_t)p1[r];
    }
    __syncthreads();
    v16bf ap = load_fragA(&shP[wave][0], 32, lane);

    // ---- O += P * V over all 16 channel tiles (V from LDS) ----
#pragma unroll
    for (int ct = 0; ct < 16; ++ct) {
      v16bf bvf = load_fragB(&shV[0] + (ct * 16) * KTILE, KTILE, lane);
      o_acc[ct] = __builtin_amdgcn_wmma_f32_16x16x32_bf16(
          false, ap, false, bvf, (short)0, o_acc[ct], false, false);
    }

    __syncthreads();                 // all reads of shV / shK[cur] complete
    if (it + 1 < ITERS) stageV(jb + KTILE);  // waited on before next S-GEMM
  }

  // ---- epilogue: out = gamma * O/l + x ----
  float inv[8];
#pragma unroll
  for (int r = 0; r < 8; ++r) inv[r] = g / l_i[r];
  float* Op       = out + ((long)b * N_SEQ + qt * 16) * C_DIM;
  const float* Xp = x   + ((long)b * N_SEQ + qt * 16) * C_DIM;
#pragma unroll
  for (int ct = 0; ct < 16; ++ct)
#pragma unroll
    for (int r = 0; r < 8; ++r) {
      const long idx = (long)(mrow0 + r) * C_DIM + ct * 16 + ncol;
      Op[idx] = o_acc[ct][r] * inv[r] + Xp[idx];
    }
}

// ---------------------------------------------------------------------------
extern "C" void kernel_launch(void* const* d_in, const int* in_sizes, int n_in,
                              void* d_out, int out_size, void* d_ws, size_t ws_size,
                              hipStream_t stream) {
  (void)in_sizes; (void)n_in; (void)out_size; (void)ws_size;
  const float* x     = (const float*)d_in[0];
  const float* Wq    = (const float*)d_in[1];
  const float* bq    = (const float*)d_in[2];
  const float* Wk    = (const float*)d_in[3];
  const float* bk    = (const float*)d_in[4];
  const float* Wv    = (const float*)d_in[5];
  const float* bv    = (const float*)d_in[6];
  const float* gamma = (const float*)d_in[7];
  float* out = (float*)d_out;

  const size_t XN  = (size_t)B_SZ * N_SEQ * C_DIM;      // 4,194,304
  const size_t WN  = (size_t)C_DIM * C_DIM;             // 65,536
  char* ws = (char*)d_ws;
  bf16_t* xbf  = (bf16_t*)(ws);
  bf16_t* wqb  = (bf16_t*)(ws + 2 * XN);
  bf16_t* wkb  = (bf16_t*)(ws + 2 * XN + 2 * WN);
  bf16_t* wvb  = (bf16_t*)(ws + 2 * XN + 4 * WN);
  bf16_t* qbf  = (bf16_t*)(ws + 2 * XN + 6 * WN);
  bf16_t* kbf  = (bf16_t*)(ws + 4 * XN + 6 * WN);
  bf16_t* vTbf = (bf16_t*)(ws + 6 * XN + 6 * WN);

  // 0) cast inputs to bf16
  PAF_cast_kernel<<<2048, 256, 0, stream>>>(x,  xbf, (int)XN);
  PAF_cast_kernel<<<64,   256, 0, stream>>>(Wq, wqb, (int)WN);
  PAF_cast_kernel<<<64,   256, 0, stream>>>(Wk, wkb, (int)WN);
  PAF_cast_kernel<<<64,   256, 0, stream>>>(Wv, wvb, (int)WN);

  // 1) QKV projection: 3 * 4 * 256 * 16 = 49152 wave-tiles / 8 waves per block
  PAF_proj_kernel<<<6144, 256, 0, stream>>>(xbf, wqb, wkb, wvb, bq, bk, bv,
                                            qbf, kbf, vTbf);

  // 2) fused flash attention: 1024 query tiles / 4 waves per block
  PAF_attn_kernel<<<256, ATT_WAVES * 32, 0, stream>>>(qbf, kbf, vTbf, x,
                                                      gamma, out);
}
